// Process_x_44564580663567
// MI455X (gfx1250) — matrix operand order repaired and checked by reading, not compiled, not run on hardware
//
#include <hip/hip_runtime.h>
#include <hip/hip_bf16.h>
#include <cstdint>

// ---- problem constants ----
#define BB   4
#define SS   4096
#define DD   1024
#define HH   16
#define DHH  64
#define WW   256
#define HIDD 2048
#define BSZ  (BB*SS)      // 16384 rows
#define LDQ  3072         // fused qkv row stride

typedef __attribute__((ext_vector_type(16))) __bf16 v16bf;
typedef __attribute__((ext_vector_type(8)))  float  v8f;
typedef __attribute__((ext_vector_type(4)))  unsigned int v4u;
typedef __attribute__((ext_vector_type(8)))  int    v8i_;
typedef __attribute__((ext_vector_type(4)))  int    v4i_;
typedef unsigned short u16;

#define WMMA_BF16(A, Bf, C) \
  __builtin_amdgcn_wmma_f32_16x16x32_bf16(false, (A), false, (Bf), (short)0, (C), false, false)

// --- Tensor Data Mover (CDNA5): 2-D bf16 tile -> LDS, guarded by probe ---
#if defined(__gfx1250__) && __has_builtin(__builtin_amdgcn_tensor_load_to_lds)
#define USE_TDM 1
#else
#define USE_TDM 0
#endif

__device__ __forceinline__ unsigned lds_u32(const void* p) {
  // generic pointer to LDS: low 32 bits are the LDS byte offset
  return (unsigned)(uintptr_t)p;
}

#if USE_TDM
__device__ __forceinline__ void tdm_load_2d(const void* gptr, unsigned lds_off,
                                            unsigned dim0 /*elems/row*/,
                                            unsigned rows, unsigned stride0 /*elems*/) {
  unsigned long long ga = (unsigned long long)(uintptr_t)gptr;
  // D# group0: count=1 | lds_addr | global_addr[56:0] | type=2 (bits 127:126)
  v4u g0 = { 1u, lds_off, (unsigned)ga, (unsigned)(ga >> 32) | 0x80000000u };
  // D# group1: data_size=2B; tensor_dim0/1; tile_dim0/1; dim0 stride
  v8i_ g1 = { (int)(1u << 16),          // workgroup_mask=0, data_size=1 (2 bytes)
              (int)(dim0 << 16),        // tensor_dim0[15:0] @ bits 63:48
              (int)(rows << 16),        // tensor_dim1[15:0] @ bits 111:80 -> dw2[31:16]
              (int)(dim0 << 16),        // tile_dim0 @ bits 127:112
              (int)rows,                // tile_dim1 @ bits 143:128
              (int)stride0,             // tensor_dim0_stride[31:0] @ bits 191:160
              0, 0 };
  v4i_ z4 = { 0, 0, 0, 0 };
  v8i_ z8 = { 0, 0, 0, 0, 0, 0, 0, 0 };
  // clang-23 / amdgpu-toolchain form: 6 args (g0, g1, g2, g3, extra, cpol)
  __builtin_amdgcn_tensor_load_to_lds(g0, g1, z4, z4, z8, 0);
}
__device__ __forceinline__ void tdm_wait() {
#if __has_builtin(__builtin_amdgcn_s_wait_tensorcnt)
  __builtin_amdgcn_s_wait_tensorcnt(0);
#else
  asm volatile("s_wait_tensorcnt 0x0" ::: "memory");
#endif
}
#endif

__device__ __forceinline__ u16 f2bf(float f) {
  unsigned int u = __float_as_uint(f);
  u += 0x7FFFu + ((u >> 16) & 1u);          // round-to-nearest-even
  return (u16)(u >> 16);
}

__device__ __forceinline__ v16bf ld16(const u16* p) {
  union { v16bf v; uint4 q[2]; } u;
  u.q[0] = *(const uint4*)(p);
  u.q[1] = *(const uint4*)(p + 8);
  return u.v;
}
__device__ __forceinline__ v16bf ld8x2(const u16* p0, const u16* p1) {
  union { v16bf v; uint4 q[2]; } u;
  u.q[0] = *(const uint4*)(p0);
  u.q[1] = *(const uint4*)(p1);
  return u.v;
}

// ---------------- fp32 -> bf16 convert ----------------
__global__ __launch_bounds__(256) void cvt_kernel(const float* __restrict__ in,
                                                  u16* __restrict__ out, int n) {
  int i = blockIdx.x * 256 + threadIdx.x;
  if (i < n) out[i] = f2bf(in[i]);
}

// Wq|Wk|Wv -> single [D][3D] bf16 buffer (column-blocked)
__global__ __launch_bounds__(256) void cvt3_kernel(const float* __restrict__ Wq,
                                                   const float* __restrict__ Wk,
                                                   const float* __restrict__ Wv,
                                                   u16* __restrict__ out) {
  int i = blockIdx.x * 256 + threadIdx.x;
  if (i < DD * DD) {
    int r = i >> 10, c = i & 1023;
    size_t o = (size_t)r * LDQ;
    out[o + c]          = f2bf(Wq[i]);
    out[o + DD + c]     = f2bf(Wk[i]);
    out[o + 2 * DD + c] = f2bf(Wv[i]);
  }
}

// ---------------- LayerNorm (row of 1024) -> bf16 ----------------
__global__ __launch_bounds__(256) void ln_kernel(const float* __restrict__ x,
                                                 const float* __restrict__ g,
                                                 const float* __restrict__ bta,
                                                 u16* __restrict__ out) {
  const int row  = blockIdx.x;
  const int tid  = threadIdx.x;
  const int lane = tid & 31, wv = tid >> 5;
  const float* xr = x + (size_t)row * DD;
  float4 v = ((const float4*)xr)[tid];              // 256*4 = 1024
  float s  = v.x + v.y + v.z + v.w;
  float s2 = v.x*v.x + v.y*v.y + v.z*v.z + v.w*v.w;
  for (int m = 16; m > 0; m >>= 1) {
    s  += __shfl_xor(s,  m, 32);
    s2 += __shfl_xor(s2, m, 32);
  }
  __shared__ float red[16];
  if (lane == 0) { red[wv] = s; red[8 + wv] = s2; }
  __syncthreads();
  float tot = 0.f, tot2 = 0.f;
  for (int i = 0; i < 8; ++i) { tot += red[i]; tot2 += red[8 + i]; }
  const float mu  = tot * (1.0f / DD);
  const float var = tot2 * (1.0f / DD) - mu * mu;
  const float rs  = rsqrtf(var + 1e-5f);
  const int c = tid * 4;
  u16* orow = out + (size_t)row * DD + c;
  const float xv[4] = {v.x, v.y, v.z, v.w};
  for (int j = 0; j < 4; ++j)
    orow[j] = f2bf((xv[j] - mu) * rs * g[c + j] + bta[c + j]);
}

// ---------------- bf16 GEMM, 128x128x32 tiles ----------------
// EPI 0: C (bf16) = acc       EPI 1: C (f32) = acc + bias[n] + res[m,n]
template <int EPI>
__global__ __launch_bounds__(256) void gemm_kernel(
    const u16* __restrict__ A, const u16* __restrict__ Bw, void* __restrict__ Cout,
    const float* __restrict__ bias, const float* __restrict__ res,
    int M, int N, int K) {
  __shared__ alignas(32) u16 As[128 * 32];
  __shared__ alignas(32) u16 Bst[128 * 32];        // transposed: [n][k]
  const int tid = threadIdx.x, lane = tid & 31, wv = tid >> 5;
  const int wm = wv & 3, wn = wv >> 2;             // 4 x 2 wave grid
  const int l15 = lane & 15;
  const int rofs = (lane >> 4) * 8;                // C row-half
  const int kofs = (lane >> 4) * 16;               // B-frag K-half
  const int m0 = blockIdx.y * 128, n0 = blockIdx.x * 128;

  v8f acc[2][4];
  for (int i = 0; i < 2; ++i)
    for (int j = 0; j < 4; ++j)
      acc[i][j] = (v8f){0.f,0.f,0.f,0.f,0.f,0.f,0.f,0.f};

  const int arow = tid >> 1, akp = (tid & 1) * 16;
  const int bk = tid >> 3,  bnp = (tid & 7) * 16;

  for (int k0 = 0; k0 < K; k0 += 32) {
    if (k0 + 32 < K) {                             // global_prefetch next K tile
      __builtin_prefetch(A  + (size_t)(m0 + arow) * K + k0 + 32, 0, 1);
      __builtin_prefetch(Bw + (size_t)(k0 + 32 + bk) * N + n0 + bnp, 0, 1);
    }
#if USE_TDM
    if (wv == 0)                                   // TDM: A tile 128x32 -> LDS
      tdm_load_2d(A + (size_t)m0 * K + k0, lds_u32(As), 32, 128, (unsigned)K);
#else
    { // stage A [128][32]
      const u16* ap = A + (size_t)(m0 + arow) * K + k0 + akp;
      *(uint4*)&As[arow * 32 + akp]     = *(const uint4*)ap;
      *(uint4*)&As[arow * 32 + akp + 8] = *(const uint4*)(ap + 8);
    }
#endif
    { // stage B transposed [128n][32k]
      const u16* bp = Bw + (size_t)(k0 + bk) * N + n0 + bnp;
      union { uint4 q; u16 u[8]; } t0, t1;
      t0.q = *(const uint4*)bp; t1.q = *(const uint4*)(bp + 8);
      for (int j = 0; j < 8; ++j) {
        Bst[(bnp + j)     * 32 + bk] = t0.u[j];
        Bst[(bnp + 8 + j) * 32 + bk] = t1.u[j];
      }
    }
#if USE_TDM
    if (wv == 0) tdm_wait();
#endif
    __syncthreads();
    for (int mt = 0; mt < 2; ++mt) {
      const u16* ab = &As[(wm * 32 + mt * 16 + l15) * 32 + kofs];
      v16bf af = ld16(ab);
      for (int nt = 0; nt < 4; ++nt) {
        v16bf bf_ = ld16(&Bst[(wn * 64 + nt * 16 + l15) * 32 + kofs]);
        acc[mt][nt] = WMMA_BF16(af, bf_, acc[mt][nt]);
      }
    }
    __syncthreads();
  }
  for (int mt = 0; mt < 2; ++mt)
    for (int nt = 0; nt < 4; ++nt) {
      const int cn = n0 + wn * 64 + nt * 16 + l15;
      for (int r = 0; r < 8; ++r) {
        const int cm = m0 + wm * 32 + mt * 16 + rofs + r;
        float vsum = acc[mt][nt][r];
        if (EPI == 0) {
          ((u16*)Cout)[(size_t)cm * N + cn] = f2bf(vsum);
        } else {
          ((float*)Cout)[(size_t)cm * N + cn] =
              vsum + bias[cn] + res[(size_t)cm * N + cn];
        }
      }
    }
}

// ---------------- GEGLU GEMM: hg = (x@W1a+b1a) * gelu(x@W1g+b1g) ----------------
__global__ __launch_bounds__(256) void geglu_kernel(
    const u16* __restrict__ A, const u16* __restrict__ W1b,
    const float* __restrict__ b1, u16* __restrict__ hg) {
  __shared__ alignas(32) u16 As[128 * 32];
  __shared__ alignas(32) u16 Ba[64 * 32];
  __shared__ alignas(32) u16 Bg[64 * 32];
  const int Nfull = 2 * HIDD;                       // 4096
  const int tid = threadIdx.x, lane = tid & 31, wv = tid >> 5;
  const int wm = wv & 3, wn = wv >> 2;
  const int l15 = lane & 15;
  const int rofs = (lane >> 4) * 8, kofs = (lane >> 4) * 16;
  const int m0 = blockIdx.y * 128, n0 = blockIdx.x * 64;

  v8f aa[2][2], ag[2][2];
  for (int i = 0; i < 2; ++i)
    for (int j = 0; j < 2; ++j) {
      aa[i][j] = (v8f){0.f,0.f,0.f,0.f,0.f,0.f,0.f,0.f};
      ag[i][j] = (v8f){0.f,0.f,0.f,0.f,0.f,0.f,0.f,0.f};
    }

  const int arow = tid >> 1, akp = (tid & 1) * 16;
  const int bk = tid >> 3,  bnp = (tid & 7) * 8;

  for (int k0 = 0; k0 < DD; k0 += 32) {
#if USE_TDM
    if (wv == 0)
      tdm_load_2d(A + (size_t)m0 * DD + k0, lds_u32(As), 32, 128, (unsigned)DD);
#else
    {
      const u16* ap = A + (size_t)(m0 + arow) * DD + k0 + akp;
      *(uint4*)&As[arow * 32 + akp]     = *(const uint4*)ap;
      *(uint4*)&As[arow * 32 + akp + 8] = *(const uint4*)(ap + 8);
    }
#endif
    {
      const u16* bp = W1b + (size_t)(k0 + bk) * Nfull + n0 + bnp;
      union { uint4 q; u16 u[8]; } ta, tg;
      ta.q = *(const uint4*)bp;
      tg.q = *(const uint4*)(bp + HIDD);
      for (int j = 0; j < 8; ++j) {
        Ba[(bnp + j) * 32 + bk] = ta.u[j];
        Bg[(bnp + j) * 32 + bk] = tg.u[j];
      }
    }
#if USE_TDM
    if (wv == 0) tdm_wait();
#endif
    __syncthreads();
    for (int mt = 0; mt < 2; ++mt) {
      const u16* ab = &As[(wm * 32 + mt * 16 + l15) * 32 + kofs];
      v16bf af = ld16(ab);
      for (int nt = 0; nt < 2; ++nt) {
        v16bf bfa = ld16(&Ba[(wn * 32 + nt * 16 + l15) * 32 + kofs]);
        v16bf bfg = ld16(&Bg[(wn * 32 + nt * 16 + l15) * 32 + kofs]);
        aa[mt][nt] = WMMA_BF16(af, bfa, aa[mt][nt]);
        ag[mt][nt] = WMMA_BF16(af, bfg, ag[mt][nt]);
      }
    }
    __syncthreads();
  }
  for (int mt = 0; mt < 2; ++mt)
    for (int nt = 0; nt < 2; ++nt) {
      const int cn = n0 + wn * 32 + nt * 16 + l15;
      for (int r = 0; r < 8; ++r) {
        const int cm = m0 + wm * 32 + mt * 16 + rofs + r;
        float av = aa[mt][nt][r] + b1[cn];
        float gv = ag[mt][nt][r] + b1[HIDD + cn];
        float ge = 0.5f * gv * (1.0f + erff(gv * 0.70710678118654752f));
        hg[(size_t)cm * HIDD + cn] = f2bf(av * ge);
      }
    }
}

// ---------------- windowed causal flash attention ----------------
// block = 128 threads (4 waves); one block per (b, h, window, 64-query-row slab)
__global__ __launch_bounds__(128) void attn_kernel(
    const u16* __restrict__ qg, const u16* __restrict__ kg,
    const u16* __restrict__ vg, u16* __restrict__ og, int ldq) {
  __shared__ alignas(32) u16 Ks[64 * 64];       // [key][d]
  __shared__ alignas(32) u16 Vt[64 * 64];       // [d][key] (transposed)
  __shared__ alignas(32) u16 Pb[4 * 16 * 64];   // per-wave P staging

  const int tid = threadIdx.x, lane = tid & 31, wv = tid >> 5;
  const int l15 = lane & 15;
  const int rofs = (lane >> 4) * 8;             // C row half
  const int kofs = (lane >> 4) * 16;            // B K half

  int bid = blockIdx.x;
  const int qb = bid & 3;  bid >>= 2;
  const int n  = bid & 15; bid >>= 4;
  const int h  = bid & 15; bid >>= 4;
  const int b  = bid;
  const int qbase = qb * 64;
  const int srow0 = n * WW + qbase + wv * 16;

  // Q fragments (K-dim = DH = 64 -> 2 steps)
  v16bf aq[2];
  {
    const u16* qp = qg + ((size_t)(b * SS + srow0 + l15)) * ldq + h * DHH;
    for (int ds = 0; ds < 2; ++ds) {
      const u16* p0 = qp + ds * 32 + kofs;
      aq[ds] = ld16(p0);
    }
  }

  float mrow[8], lrow[8];
  v8f o[4];
  for (int r = 0; r < 8; ++r) { mrow[r] = -3.0e38f; lrow[r] = 0.f; }
  for (int t = 0; t < 4; ++t) o[t] = (v8f){0.f,0.f,0.f,0.f,0.f,0.f,0.f,0.f};

  const int kb_start = (n == 0) ? 4 : 0;
  const int kb_end   = 4 + qb;          // diagonal block

  for (int kb = kb_start; kb <= kb_end; ++kb) {
    __syncthreads();
    { // stage K (TDM if available) and V (transposed, VGPR path)
      const int s0 = (n - 1) * WW + kb * 64;
#if USE_TDM
      if (wv == 0)
        tdm_load_2d(kg + ((size_t)(b * SS + s0)) * ldq + h * DHH, lds_u32(Ks),
                    64, 64, (unsigned)ldq);
#endif
      for (int i = 0; i < 4; ++i) {
        const int c = tid + i * 128;              // 0..511 chunks of 8
        const int row = c >> 3, d8 = (c & 7) * 8;
        const size_t gofs = ((size_t)(b * SS + s0 + row)) * ldq + h * DHH + d8;
#if !USE_TDM
        *(uint4*)&Ks[row * 64 + d8] = *(const uint4*)(kg + gofs);
#endif
        union { uint4 q; u16 u[8]; } vvv;
        vvv.q = *(const uint4*)(vg + gofs);
        for (int j = 0; j < 8; ++j) Vt[(d8 + j) * 64 + row] = vvv.u[j];
      }
#if USE_TDM
      if (wv == 0) tdm_wait();
#endif
    }
    __syncthreads();

    // scores: S = Q * K^T  (B operand = K rows, contiguous in LDS)
    v8f s[4];
    for (int nt = 0; nt < 4; ++nt) {
      v8f z = (v8f){0.f,0.f,0.f,0.f,0.f,0.f,0.f,0.f};
      for (int ds = 0; ds < 2; ++ds) {
        v16bf bkf = ld16(&Ks[(nt * 16 + l15) * 64 + ds * 32 + kofs]);
        z = WMMA_BF16(aq[ds], bkf, z);
      }
      s[nt] = z;
    }
    const bool diag = (kb == kb_end);
    for (int nt = 0; nt < 4; ++nt) {
      const int col = nt * 16 + l15;
      for (int r = 0; r < 8; ++r) {
        float sv = s[nt][r] * 0.125f;             // DH^-0.5
        if (diag && col > (wv * 16 + rofs + r)) sv = -3.0e38f;
        s[nt][r] = sv;
      }
    }
    // per-row max across 64 keys (rows split by lane half; reduce 16-lane groups)
    float mnew[8], scl[8];
    for (int r = 0; r < 8; ++r) {
      float v0 = fmaxf(fmaxf(s[0][r], s[1][r]), fmaxf(s[2][r], s[3][r]));
      for (int m = 1; m < 16; m <<= 1) v0 = fmaxf(v0, __shfl_xor(v0, m, 32));
      mnew[r] = fmaxf(mrow[r], v0);
      scl[r]  = expf(mrow[r] - mnew[r]);
      mrow[r] = mnew[r];
    }
    // P = exp(S - m), stage to per-wave LDS in A-layout source form
    float ps[8];
    for (int r = 0; r < 8; ++r) ps[r] = 0.f;
    u16* pw = &Pb[wv * 16 * 64];
    for (int nt = 0; nt < 4; ++nt) {
      const int col = nt * 16 + l15;
      for (int r = 0; r < 8; ++r) {
        float p = expf(s[nt][r] - mnew[r]);
        ps[r] += p;
        pw[(rofs + r) * 64 + col] = f2bf(p);
      }
    }
    for (int r = 0; r < 8; ++r) {
      float v0 = ps[r];
      for (int m = 1; m < 16; m <<= 1) v0 += __shfl_xor(v0, m, 32);
      lrow[r] = lrow[r] * scl[r] + v0;
      for (int t = 0; t < 4; ++t) o[t][r] *= scl[r];
    }
    // O += P * V  (per-wave LDS staging; same-wave DS ordering is in-order)
    for (int ks = 0; ks < 2; ++ks) {
      const u16* pp = pw + (size_t)l15 * 64 + ks * 32 + kofs;
      v16bf ap = ld16(pp);
      for (int t = 0; t < 4; ++t) {
        v16bf bvf = ld16(&Vt[(t * 16 + l15) * 64 + ks * 32 + kofs]);
        o[t] = WMMA_BF16(ap, bvf, o[t]);
      }
    }
  }

  // normalize and store
  for (int t = 0; t < 4; ++t) {
    const int dh = t * 16 + l15;
    for (int r = 0; r < 8; ++r) {
      const int sidx = srow0 + rofs + r;
      og[((size_t)(b * SS + sidx)) * DD + h * DHH + dh] = f2bf(o[t][r] / lrow[r]);
    }
  }
}

// ---------------- host launcher ----------------
extern "C" void kernel_launch(void* const* d_in, const int* in_sizes, int n_in,
                              void* d_out, int out_size, void* d_ws, size_t ws_size,
                              hipStream_t stream) {
  const float* x    = (const float*)d_in[0];
  const float* ln1g = (const float*)d_in[1];
  const float* ln1b = (const float*)d_in[2];
  const float* ln2g = (const float*)d_in[3];
  const float* ln2b = (const float*)d_in[4];
  const float* Wq   = (const float*)d_in[5];
  const float* Wk   = (const float*)d_in[6];
  const float* Wv   = (const float*)d_in[7];
  const float* Wo   = (const float*)d_in[8];
  const float* bo   = (const float*)d_in[9];
  const float* W1   = (const float*)d_in[10];
  const float* b1   = (const float*)d_in[11];
  const float* W2   = (const float*)d_in[12];
  const float* b2   = (const float*)d_in[13];

  char* ws = (char*)d_ws;
  size_t off = 0;
  auto alloc = [&](size_t bytes) {
    char* p = ws + off;
    off += (bytes + 255) & ~(size_t)255;
    return p;
  };
  u16*   d_wqkv = (u16*)alloc((size_t)DD * LDQ * 2);   // 6 MB
  u16*   d_wo   = (u16*)alloc((size_t)DD * DD * 2);    // 2 MB
  u16*   d_w1   = (u16*)alloc((size_t)DD * 2 * HIDD * 2); // 8 MB
  u16*   d_w2   = (u16*)alloc((size_t)HIDD * DD * 2);  // 4 MB
  u16*   d_xn   = (u16*)alloc((size_t)BSZ * DD * 2);   // 32 MB (LN1 out, reused for LN2)
  u16*   d_qkv  = (u16*)alloc((size_t)BSZ * LDQ * 2);  // 96 MB
  u16*   d_ao   = (u16*)alloc((size_t)BSZ * DD * 2);   // 32 MB
  float* d_y1   = (float*)alloc((size_t)BSZ * DD * 4); // 64 MB
  u16*   d_hg   = d_qkv;                               // alias (64 MB needed)

  // weights -> bf16
  cvt3_kernel<<<(DD*DD + 255) / 256, 256, 0, stream>>>(Wq, Wk, Wv, d_wqkv);
  cvt_kernel<<<(DD*DD + 255) / 256, 256, 0, stream>>>(Wo, d_wo, DD * DD);
  cvt_kernel<<<(DD*2*HIDD + 255) / 256, 256, 0, stream>>>(W1, d_w1, DD * 2 * HIDD);
  cvt_kernel<<<(HIDD*DD + 255) / 256, 256, 0, stream>>>(W2, d_w2, HIDD * DD);

  // LN1
  ln_kernel<<<BSZ, 256, 0, stream>>>(x, ln1g, ln1b, d_xn);

  // fused QKV projection: [BS,1024] x [1024,3072]
  dim3 gqkv(LDQ / 128, BSZ / 128);
  gemm_kernel<0><<<gqkv, 256, 0, stream>>>(d_xn, d_wqkv, d_qkv, nullptr, nullptr,
                                           BSZ, LDQ, DD);

  // windowed causal attention (q/k/v are column slices of qkv, stride LDQ)
  attn_kernel<<<BB * HH * (SS / WW) * (WW / 64), 128, 0, stream>>>(
      d_qkv, d_qkv + DD, d_qkv + 2 * DD, d_ao, LDQ);

  // O-projection + bias + residual -> y1 (f32)
  dim3 gq(DD / 128, BSZ / 128);
  gemm_kernel<1><<<gq, 256, 0, stream>>>(d_ao, d_wo, d_y1, bo, x, BSZ, DD, DD);

  // LN2 (reuse xn)
  ln_kernel<<<BSZ, 256, 0, stream>>>(d_y1, ln2g, ln2b, d_xn);

  // GEGLU: hg = a * gelu(g)
  dim3 gg(HIDD / 64, BSZ / 128);
  geglu_kernel<<<gg, 256, 0, stream>>>(d_xn, d_w1, b1, d_hg);

  // W2 + bias + residual -> out (f32)
  gemm_kernel<1><<<gq, 256, 0, stream>>>(d_hg, d_w2, (float*)d_out, b2, d_y1,
                                         BSZ, DD, HIDD);
}
